// LSTMAutoencoder_59072980189741
// MI455X (gfx1250) — matrix operand order repaired and checked by reading, not compiled, not run on hardware
//
#include <hip/hip_runtime.h>

// ---------------- problem constants ----------------
#define NBLK  65          // 64 gate-blocks + 1 output-projection block
#define HD    1024
#define SEQ   512
#define BATCH 64
#define DIN   64
#define LAT   128

typedef unsigned short u16;
typedef __attribute__((ext_vector_type(16))) __bf16 v16bf;
typedef __attribute__((ext_vector_type(8)))  float  v8f;

struct alignas(16) U4 { unsigned x, y, z, w; };
union  ABu { U4 q[2]; v16bf v; };

// ---------------- workspace layout (bytes) ----------------
static constexpr size_t OFF_BAR  = 0;                                  // 2 uints (barrier)
static constexpr size_t OFF_HBUF = 256;                                // 2 x [64,1024] bf16
static constexpr size_t OFF_XBF  = OFF_HBUF + 2ul*BATCH*HD*2;          // [64,512,64] bf16
static constexpr size_t OFF_WIH  = OFF_XBF  + (size_t)BATCH*SEQ*DIN*2; // [4096,64] bf16
static constexpr size_t OFF_WHHE = OFF_WIH  + 4ul*HD*DIN*2;            // [4096,1024] bf16
static constexpr size_t OFF_WHHD = OFF_WHHE + 4ul*HD*HD*2;             // [4096,1024] bf16
static constexpr size_t OFF_WFUS = OFF_WHHD + 4ul*HD*HD*2;             // fused dec weight
static constexpr size_t OFF_MLAT = OFF_WFUS + 4ul*HD*HD*2;             // [1024,1024] bf16
static constexpr size_t OFF_OUTW = OFF_MLAT + (size_t)HD*HD*2;         // [64,1024] bf16
static constexpr size_t OFF_BFUS = OFF_OUTW + (size_t)DIN*HD*2;        // [4096] f32
static constexpr size_t OFF_BLAT = OFF_BFUS + 4ul*HD*4;                // [1024] f32
static constexpr size_t WS_NEED  = OFF_BLAT + (size_t)HD*4;            // ~33 MB

// ---------------- helpers ----------------
__device__ __forceinline__ u16 f2bf(float f) {
  unsigned u = __float_as_uint(f);
  unsigned r = u + 0x7FFFu + ((u >> 16) & 1u);   // round-to-nearest-even
  return (u16)(r >> 16);
}

__device__ __forceinline__ float sigm(float x) { return 1.f / (1.f + __expf(-x)); }
__device__ __forceinline__ float tanh_(float x) {
  float e = __expf(-2.f * x);
  return (1.f - e) / (1.f + e);
}

__device__ __forceinline__ v8f splat8(float v) {
  v8f r;
  #pragma unroll
  for (int i = 0; i < 8; ++i) r[i] = v;
  return r;
}

__device__ __forceinline__ v8f wmma_bf16(v16bf a, v16bf b, v8f c) {
  // D = A(16x32 bf16) * B(32x16 bf16) + C(16x16 f32)
  return __builtin_amdgcn_wmma_f32_16x16x32_bf16(false, a, false, b, (short)0, c,
                                                 false, false);
}

// A fragment (16x32, bf16): lanes 0-15 row r hold K {k0..k0+7, k0+16..k0+23},
// lanes 16-31 row r hold K {k0+8..k0+15, k0+24..k0+31}; 2 K per VGPR.
__device__ __forceinline__ v16bf loadA16(const u16* __restrict__ base, int ld,
                                         int arow, int hi, int k0) {
  const u16* p = base + (size_t)arow * ld + k0 + (hi << 3);
  ABu u;
  u.q[0] = *(const U4*)p;
  u.q[1] = *(const U4*)(p + 16);
  return u.v;
}

// B fragment (32x16, bf16) from weight stored [N,K] row-major (global or LDS):
// lane n (0-15) = column n, K {k0..k0+7, k0+16..k0+23}; lanes 16-31 K {k0+8..k0+15, k0+24..k0+31}.
__device__ __forceinline__ v16bf loadB16(const u16* W, int ld, int brow, int hi,
                                         int k0) {
  const u16* p = W + (size_t)brow * ld + k0 + (hi << 3);
  ABu u;
  u.q[0] = *(const U4*)p;
  u.q[1] = *(const U4*)(p + 16);
  return u.v;
}

// Device-wide generation barrier (state zeroed by the setup kernel each call).
__device__ __forceinline__ void gridBarrier(unsigned* bar, unsigned gen) {
  __syncthreads();
  if (threadIdx.x == 0) {
    __threadfence();
    unsigned prev = atomicAdd(&bar[0], 1u);
    if (prev == (unsigned)(NBLK - 1)) {
      atomicExch(&bar[0], 0u);
      __threadfence();
      atomicAdd(&bar[1], 1u);               // release generation gen -> gen+1
    } else {
      while (__atomic_load_n(&bar[1], __ATOMIC_RELAXED) <= gen)
        __builtin_amdgcn_s_sleep(1);
    }
    __threadfence();
  }
  __syncthreads();
}

// ---------------- setup kernel: bf16 conversion + weight fusion ----------------
__global__ void __launch_bounds__(256) ae_setup(
    const float* __restrict__ x,
    const float* __restrict__ eWih, const float* __restrict__ eWhh,
    const float* __restrict__ efcW, const float* __restrict__ efcb,
    const float* __restrict__ dfcW, const float* __restrict__ dfcb,
    const float* __restrict__ dWih, const float* __restrict__ dWhh,
    const float* __restrict__ db,   const float* __restrict__ oW,
    const float* __restrict__ ob,   unsigned char* __restrict__ ws)
{
  const int tid = blockIdx.x * blockDim.x + threadIdx.x;
  const int T   = gridDim.x * blockDim.x;
  u16* xbf  = (u16*)(ws + OFF_XBF);
  u16* wih  = (u16*)(ws + OFF_WIH);
  u16* whhe = (u16*)(ws + OFF_WHHE);
  u16* whhd = (u16*)(ws + OFF_WHHD);
  u16* wfus = (u16*)(ws + OFF_WFUS);
  u16* mlat = (u16*)(ws + OFF_MLAT);
  u16* outw = (u16*)(ws + OFF_OUTW);
  float* bfus = (float*)(ws + OFF_BFUS);
  float* blat = (float*)(ws + OFF_BLAT);

  for (int i = tid; i < BATCH*SEQ*DIN; i += T) xbf[i]  = f2bf(x[i]);
  for (int i = tid; i < 4*HD*DIN;      i += T) wih[i]  = f2bf(eWih[i]);
  for (int i = tid; i < 4*HD*HD;       i += T) whhe[i] = f2bf(eWhh[i]);
  for (int i = tid; i < 4*HD*HD;       i += T) whhd[i] = f2bf(dWhh[i]);
  for (int i = tid; i < DIN*HD;        i += T) outw[i] = f2bf(oW[i]);

  // W' = dec_Whh + dec_Wih @ outW  (folds output->input feedback into recurrence)
  for (int i = tid; i < 4*HD*HD; i += T) {
    int n = i >> 10, k = i & (HD - 1);
    float acc = dWhh[i];
    const float* wr = dWih + n * DIN;
    #pragma unroll 8
    for (int d = 0; d < DIN; ++d) acc = fmaf(wr[d], oW[d*HD + k], acc);
    wfus[i] = f2bf(acc);
  }
  // M' = dec_fcW @ enc_fcW  (fused latent bottleneck transition)
  for (int i = tid; i < HD*HD; i += T) {
    int n = i >> 10, k = i & (HD - 1);
    float acc = 0.f;
    #pragma unroll 8
    for (int l = 0; l < LAT; ++l) acc = fmaf(dfcW[n*LAT + l], efcW[l*HD + k], acc);
    mlat[i] = f2bf(acc);
  }
  // b' = dec_b + dec_Wih @ outb
  for (int i = tid; i < 4*HD; i += T) {
    float acc = db[i];
    for (int d = 0; d < DIN; ++d) acc = fmaf(dWih[i*DIN + d], ob[d], acc);
    bfus[i] = acc;
  }
  // b'' = dec_fcb + dec_fcW @ enc_fcb
  for (int i = tid; i < HD; i += T) {
    float acc = dfcb[i];
    for (int l = 0; l < LAT; ++l) acc = fmaf(dfcW[i*LAT + l], efcb[l], acc);
    blat[i] = acc;
  }
  // zero both h buffers (h0 = 0) and the barrier state
  unsigned* hz = (unsigned*)(ws + OFF_HBUF);
  for (int i = tid; i < (2*BATCH*HD) / 2; i += T) hz[i] = 0u;
  if (tid == 0) { unsigned* bar = (unsigned*)(ws + OFF_BAR); bar[0] = 0u; bar[1] = 0u; }
}

// ---------------- LDS weight staging ----------------
// Copy this block's 64 weight rows ([64][1024] bf16 = 128 KB) into LDS.
// gateMap=true : local row lr = g*16+c  ->  global row g*1024 + j0 + c
// gateMap=false: local row lr          ->  global row lr      (out-projection)
__device__ __forceinline__ void fill_lds_w(u16* dst, const u16* __restrict__ src,
                                           int j0, bool gateMap) {
  for (int ch = threadIdx.x; ch < 64 * 128; ch += 256) {   // 16-byte chunks
    int lr = ch >> 7;
    int of = (ch & 127) << 3;
    int gr = gateMap ? ((lr >> 4) * HD + j0 + (lr & 15)) : lr;
    *(U4*)(dst + lr * HD + of) = *(const U4*)(src + (size_t)gr * HD + of);
  }
}
// enc_Wih slice: [64][64] bf16 = 8 KB
__device__ __forceinline__ void fill_lds_wih(u16* dst, const u16* __restrict__ src,
                                             int j0) {
  for (int ch = threadIdx.x; ch < 64 * 8; ch += 256) {
    int lr = ch >> 3;
    int of = (ch & 7) << 3;
    int gr = (lr >> 4) * HD + j0 + (lr & 15);
    *(U4*)(dst + lr * DIN + of) = *(const U4*)(src + (size_t)gr * DIN + of);
  }
}

// ---------------- per-step building blocks ----------------
// One LSTM step for a gate-block owning 16 h-columns (j0..j0+15).
// Waves 0-3: (i,f) gates for M-tiles 0-3; waves 4-7: (g,o). c stays in registers
// of waves 0-3 for the entire sequence. B fragments come from LDS (BLDS=true,
// weight slice pre-staged) or global (decoder step 0 only).
template <bool BLDS>
__device__ __forceinline__ void lstm_gate_step(
    const u16* __restrict__ hp, u16* __restrict__ hn,
    const u16* __restrict__ Wg, const u16* Wl,
    const float* __restrict__ bias,
    const u16* __restrict__ xA, const u16* xWl, int t,
    int j0, int w, int lane, float* creg,
    float (&ldsG)[4][16][17], float (&ldsO)[4][16][17])
{
  const int m = w & 3, row0 = m * 16;
  const int hi = lane >> 4, col = lane & 15;
  const int arow = row0 + col;
  const int g0 = (w < 4) ? 0 : 2;                      // (i,f) or (g,o)
  const int n0 = g0 * HD + j0, n1 = (g0 + 1) * HD + j0;
  const int lb0 = g0 * 16 + col, lb1 = (g0 + 1) * 16 + col;   // LDS-local rows
  v8f acc0 = splat8(bias[n0 + col]);
  v8f acc1 = splat8(bias[n1 + col]);

  if (xA) {                                            // encoder: + x_t @ Wih^T (K=64)
    #pragma unroll
    for (int k = 0; k < DIN; k += 32) {
      v16bf a = loadA16(xA + (size_t)t * DIN, SEQ * DIN, arow, hi, k);
      acc0 = wmma_bf16(a, loadB16(xWl, DIN, lb0, hi, k), acc0);
      acc1 = wmma_bf16(a, loadB16(xWl, DIN, lb1, hi, k), acc1);
    }
  }
  #pragma unroll 4
  for (int k = 0; k < HD; k += 32) {                   // + h @ W^T (K=1024)
    v16bf a = loadA16(hp, HD, arow, hi, k);
    if (BLDS) {
      acc0 = wmma_bf16(a, loadB16(Wl, HD, lb0, hi, k), acc0);
      acc1 = wmma_bf16(a, loadB16(Wl, HD, lb1, hi, k), acc1);
    } else {
      acc0 = wmma_bf16(a, loadB16(Wg, HD, n0 + col, hi, k), acc0);
      acc1 = wmma_bf16(a, loadB16(Wg, HD, n1 + col, hi, k), acc1);
    }
  }

  if (w >= 4) {                                        // publish g,o tiles
    #pragma unroll
    for (int r = 0; r < 8; ++r) {
      ldsG[m][r + hi * 8][col] = acc0[r];
      ldsO[m][r + hi * 8][col] = acc1[r];
    }
  }
  __syncthreads();
  if (w < 4) {                                         // LSTM cell update + h store
    #pragma unroll
    for (int r = 0; r < 8; ++r) {
      int rr = r + hi * 8;
      float iv = sigm(acc0[r]);
      float fv = sigm(acc1[r]);
      float gv = tanh_(ldsG[m][rr][col]);
      float ov = sigm(ldsO[m][rr][col]);
      float c  = fmaf(fv, creg[r], iv * gv);
      creg[r] = c;
      hn[(size_t)(row0 + rr) * HD + j0 + col] = f2bf(ov * tanh_(c));
    }
  }
  __syncthreads();
}

// Output projection out_t = h_t @ outW^T + outb (outW LDS-resident), done by the
// dedicated block one step behind the recurrence (fully overlapped).
__device__ __forceinline__ void out_proj_step(
    const u16* __restrict__ hp, const u16* Wl,
    const float* __restrict__ ob, float* __restrict__ out,
    int tout, int w, int lane)
{
  const int m = w & 3, row0 = m * 16;
  const int hi = lane >> 4, col = lane & 15;
  const int arow = row0 + col;
  const int n0 = (w >> 2) * 32, n1 = n0 + 16;
  v8f acc0 = splat8(ob[n0 + col]);
  v8f acc1 = splat8(ob[n1 + col]);
  #pragma unroll 4
  for (int k = 0; k < HD; k += 32) {
    v16bf a = loadA16(hp, HD, arow, hi, k);
    acc0 = wmma_bf16(a, loadB16(Wl, HD, n0 + col, hi, k), acc0);
    acc1 = wmma_bf16(a, loadB16(Wl, HD, n1 + col, hi, k), acc1);
  }
  #pragma unroll
  for (int r = 0; r < 8; ++r) {
    int rr = r + hi * 8;
    size_t base = (size_t)(row0 + rr) * (SEQ * DIN) + (size_t)tout * DIN;
    out[base + n0 + col] = acc0[r];
    out[base + n1 + col] = acc1[r];
  }
}

// ---------------- persistent recurrent kernel ----------------
__global__ void __launch_bounds__(256) ae_persistent(
    const float* __restrict__ enc_b, const float* __restrict__ dec_b,
    const float* __restrict__ outb,
    unsigned char* __restrict__ ws, float* __restrict__ out)
{
  // 128 KB weight slice + 8 KB x-projection slice + 8.5 KB gate exchange
  // (~145 KB of the WGP's 320 KB LDS; 2 workgroups still fit per WGP).
  __shared__ alignas(16) u16 ldsW[64 * HD];
  __shared__ alignas(16) u16 ldsWih[64 * DIN];
  __shared__ float ldsG[4][16][17];
  __shared__ float ldsO[4][16][17];

  unsigned* bar = (unsigned*)(ws + OFF_BAR);
  u16* hbuf0 = (u16*)(ws + OFF_HBUF);
  u16* hbuf1 = hbuf0 + BATCH * HD;
  const u16* xbf  = (const u16*)(ws + OFF_XBF);
  const u16* Wih  = (const u16*)(ws + OFF_WIH);
  const u16* WhhE = (const u16*)(ws + OFF_WHHE);
  const u16* WhhD = (const u16*)(ws + OFF_WHHD);
  const u16* Wfus = (const u16*)(ws + OFF_WFUS);
  const u16* Mlat = (const u16*)(ws + OFF_MLAT);
  const u16* oWb  = (const u16*)(ws + OFF_OUTW);
  const float* bfus = (const float*)(ws + OFF_BFUS);
  const float* blat = (const float*)(ws + OFF_BLAT);

  const int lane = threadIdx.x & 31;
  const int w    = threadIdx.x >> 5;
  const bool isGate = blockIdx.x < 64;
  const int j0 = (int)blockIdx.x * 16;      // h-columns owned by a gate block
  unsigned gen = 0;
  float creg[8];
  #pragma unroll
  for (int r = 0; r < 8; ++r) creg[r] = 0.f;

  // Stage phase-1 weights into LDS (gate blocks: enc_Whh slice + enc_Wih slice;
  // out block: outW, kept for the whole decoder).
  if (isGate) {
    fill_lds_w(ldsW, WhhE, j0, true);
    fill_lds_wih(ldsWih, Wih, j0);
  } else {
    fill_lds_w(ldsW, oWb, 0, false);
  }
  __syncthreads();

  // -------- encoder: 512 steps (h0 = c0 = 0) --------
  for (int t = 0; t < SEQ; ++t) {
    if (isGate) {
      const u16* hp = (t & 1) ? hbuf1 : hbuf0;
      u16*       hn = (t & 1) ? hbuf0 : hbuf1;
      lstm_gate_step<true>(hp, hn, nullptr, ldsW, enc_b, xbf, ldsWih, t,
                           j0, w, lane, creg, ldsG, ldsO);
    }
    gridBarrier(bar, gen); ++gen;
  }

  // -------- swap LDS weights to the fused decoder recurrence W' --------
  if (isGate) {
    fill_lds_w(ldsW, Wfus, j0, true);
    __syncthreads();
  }

  // -------- latent transition: h_dec = h_enc @ M'^T + b'' ; c = 0 --------
  if (isGate) {
    #pragma unroll
    for (int r = 0; r < 8; ++r) creg[r] = 0.f;
    if (w < 4) {
      const int m = w & 3, row0 = m * 16, hi = lane >> 4, col = lane & 15;
      v8f acc = splat8(blat[j0 + col]);
      #pragma unroll 4
      for (int k = 0; k < HD; k += 32) {
        v16bf a = loadA16(hbuf0, HD, row0 + col, hi, k);   // h_enc is in hbuf0
        acc = wmma_bf16(a, loadB16(Mlat, HD, j0 + col, hi, k), acc);
      }
      #pragma unroll
      for (int r = 0; r < 8; ++r)
        hbuf1[(size_t)(row0 + r + hi * 8) * HD + j0 + col] = f2bf(acc[r]);
    }
  }
  gridBarrier(bar, gen); ++gen;

  // -------- decoder: 512 steps + pipelined output projection --------
  // step 0 uses raw dec_Whh/dec_b from global (x_in = 0); steps >=1 use the
  // LDS-resident fused W'/b'.
  for (int t = 0; t <= SEQ; ++t) {
    if (isGate && t < SEQ) {
      const u16* hp = (t & 1) ? hbuf0 : hbuf1;  // h_{t-1} (h_dec lives in hbuf1)
      u16*       hn = (t & 1) ? hbuf1 : hbuf0;  // h_t
      if (t == 0)
        lstm_gate_step<false>(hp, hn, WhhD, nullptr, dec_b, nullptr, nullptr, 0,
                              j0, w, lane, creg, ldsG, ldsO);
      else
        lstm_gate_step<true>(hp, hn, nullptr, ldsW, bfus, nullptr, nullptr, 0,
                             j0, w, lane, creg, ldsG, ldsO);
    }
    if (!isGate && t >= 1) {
      const int tout = t - 1;
      const u16* hsrc = (tout & 1) ? hbuf1 : hbuf0;        // h_{tout}
      out_proj_step(hsrc, ldsW, outb, out, tout, w, lane);
    }
    gridBarrier(bar, gen); ++gen;
  }
}

// ---------------- host entry ----------------
extern "C" void kernel_launch(void* const* d_in, const int* in_sizes, int n_in,
                              void* d_out, int out_size, void* d_ws, size_t ws_size,
                              hipStream_t stream) {
  (void)in_sizes; (void)n_in; (void)out_size;
  if (ws_size < WS_NEED) return;   // workspace too small; nothing safe to do

  const float* x    = (const float*)d_in[0];
  const float* eWih = (const float*)d_in[1];
  const float* eWhh = (const float*)d_in[2];
  const float* eb   = (const float*)d_in[3];
  const float* efcW = (const float*)d_in[4];
  const float* efcb = (const float*)d_in[5];
  const float* dfcW = (const float*)d_in[6];
  const float* dfcb = (const float*)d_in[7];
  const float* dWih = (const float*)d_in[8];
  const float* dWhh = (const float*)d_in[9];
  const float* db   = (const float*)d_in[10];
  const float* oW   = (const float*)d_in[11];
  const float* ob   = (const float*)d_in[12];
  unsigned char* ws = (unsigned char*)d_ws;

  ae_setup<<<dim3(512), dim3(256), 0, stream>>>(
      x, eWih, eWhh, efcW, efcb, dfcW, dfcb, dWih, dWhh, db, oW, ob, ws);

  ae_persistent<<<dim3(NBLK), dim3(256), 0, stream>>>(
      eb, db, ob, ws, (float*)d_out);
}
